// GATMGEmbedder_8538394984945
// MI455X (gfx1250) — matrix operand order repaired and checked by reading, not compiled, not run on hardware
//
#include <hip/hip_runtime.h>
#include <hip/hip_bf16.h>

typedef __attribute__((ext_vector_type(16))) _Float16 v16h;
typedef __attribute__((ext_vector_type(8)))  _Float16 v8h;
typedef __attribute__((ext_vector_type(8)))  float    v8f;

#define NN_  40000
#define EE_  640000
#define GG_  40
#define HH_  2
#define CC_  128
#define DD_  256
#define SLOPE_ 0.2f
#define EPS_   1e-5f

__device__ __forceinline__ float lrelu(float x) { return x >= 0.f ? x : SLOPE_ * x; }

// monotone float <-> uint encoding for atomicMax on floats
__device__ __forceinline__ unsigned f2ord(float f) {
    unsigned u = __float_as_uint(f);
    return (u & 0x80000000u) ? ~u : (u | 0x80000000u);
}
__device__ __forceinline__ float ord2f(unsigned u) {
    return (u & 0x80000000u) ? __uint_as_float(u & 0x7fffffffu) : __uint_as_float(~u);
}

// ---------------- fill / copy ----------------
__global__ void fill_u32_kernel(unsigned* __restrict__ p, unsigned v, int n) {
    int i = blockIdx.x * blockDim.x + threadIdx.x;
    if (i < n) p[i] = v;
}

__global__ void copy_kernel(float* __restrict__ dst, const float* __restrict__ src, int n) {
    int i = blockIdx.x * blockDim.x + threadIdx.x;
    if (i < n) dst[i] = src[i];
}

// ---------------- packers ----------------
// row-major f32 -> row-major f16 (activations)
__global__ void pack_a_kernel(const float* __restrict__ A, _Float16* __restrict__ Ah, int n) {
    int i = blockIdx.x * blockDim.x + threadIdx.x;
    if (i < n) Ah[i] = (_Float16)A[i];
}

// W[K x Dc] f32 -> Bp[(K/32)*Dc][32] f16, where the 32 halves for (ktile, col) are
// K = ktile*32 + q in order q = 0..31  (q == lane_half*16 + fragment_element)
__global__ void pack_b_kernel(const float* __restrict__ W, _Float16* __restrict__ Bp,
                              int K, int Dc) {
    int idx = blockIdx.x * blockDim.x + threadIdx.x;
    if (idx >= K * Dc) return;
    int q   = idx & 31;
    int col = (idx >> 5) % Dc;
    int kt  = (idx >> 5) / Dc;
    Bp[idx] = (_Float16)W[(size_t)(kt * 32 + q) * Dc + col];
}

// ---------------- WMMA GEMM: Cout[M x Dc] = Ah[M x K] * Bp (f16 packed, f32 accum) ------
// one wave -> 16 rows x 64 cols (4 accumulators); all fragment loads are contiguous b128
__global__ __launch_bounds__(256) void gemm_wmma_f16p(const _Float16* __restrict__ Ah,
                                                      const _Float16* __restrict__ Bp,
                                                      float* __restrict__ Cout,
                                                      int M, int K, int Dc) {
    int wave = (blockIdx.x * blockDim.x + threadIdx.x) >> 5;
    int lane = threadIdx.x & 31;
    int nBlk = Dc >> 6;
    int totalWaves = (M >> 4) * nBlk;
    if (wave >= totalWaves) return;
    int mBase = (wave / nBlk) << 4;
    int nBase = (wave % nBlk) << 6;
    int lhalf = lane >> 4;
    int lmod  = lane & 15;

    v8f acc[4] = {};
    for (int k0 = 0; k0 < K; k0 += 32) {
        // A fragment: row = mBase+lmod; elements 0..15 -> K = k0+16*lhalf+j
        const _Float16* arow = Ah + (size_t)(mBase + lmod) * K + k0;
        v8h alo = *(const v8h*)(arow + 16 * lhalf);
        v8h ahi = *(const v8h*)(arow + 16 * lhalf + 8);
        v16h a = __builtin_shufflevector(alo, ahi, 0, 1, 2, 3, 4, 5, 6, 7,
                                         8, 9, 10, 11, 12, 13, 14, 15);
        const _Float16* bt = Bp + ((size_t)(k0 >> 5) * Dc) * 32;
#pragma unroll
        for (int t = 0; t < 4; ++t) {
            const _Float16* bc = bt + (size_t)(nBase + t * 16 + lmod) * 32 + lhalf * 16;
            v8h blo = *(const v8h*)(bc);
            v8h bhi = *(const v8h*)(bc + 8);
            v16h b = __builtin_shufflevector(blo, bhi, 0, 1, 2, 3, 4, 5, 6, 7,
                                             8, 9, 10, 11, 12, 13, 14, 15);
            acc[t] = __builtin_amdgcn_wmma_f32_16x16x32_f16(
                false, a, false, b, (short)0, acc[t], false, false);
        }
    }
#pragma unroll
    for (int t = 0; t < 4; ++t) {
#pragma unroll
        for (int v = 0; v < 8; ++v) {
            Cout[(size_t)(mBase + v + 8 * lhalf) * Dc + nBase + t * 16 + lmod] = acc[t][v];
        }
    }
}

// ---------------- attention logits: one wave per (n,hd); lane = 4-channel group ---------
__global__ void attn_kernel(const float* __restrict__ h, const float* __restrict__ al,
                            const float* __restrict__ ar, float* __restrict__ el,
                            float* __restrict__ er) {
    int wid  = (blockIdx.x * blockDim.x + threadIdx.x) >> 5;
    int lane = threadIdx.x & 31;
    if (wid >= NN_ * HH_) return;
    int n = wid >> 1, hd = wid & 1;
    const float4 v  = *(const float4*)(h + (size_t)n * DD_ + hd * CC_ + lane * 4);
    const float4 wa = *(const float4*)(al + hd * CC_ + lane * 4);
    const float4 wr = *(const float4*)(ar + hd * CC_ + lane * 4);
    float sl = v.x * wa.x + v.y * wa.y + v.z * wa.z + v.w * wa.w;
    float sr = v.x * wr.x + v.y * wr.y + v.z * wr.z + v.w * wr.w;
#pragma unroll
    for (int o = 16; o; o >>= 1) {
        sl += __shfl_xor(sl, o, 32);
        sr += __shfl_xor(sr, o, 32);
    }
    if (lane == 0) { el[wid] = sl; er[wid] = sr; }
}

// ---------------- edge pass A: e = leaky(el[src]+er[dst]); atomic max into m[dst] -------
__global__ void edge_a_kernel(const int* __restrict__ src, const int* __restrict__ dst,
                              const float* __restrict__ el, const float* __restrict__ er,
                              float* __restrict__ ebuf, unsigned* __restrict__ mbuf) {
    int idx = blockIdx.x * blockDim.x + threadIdx.x;
    if (idx >= EE_ * HH_) return;
    int e = idx >> 1, hd = idx & 1;
    float v = lrelu(el[src[e] * HH_ + hd] + er[dst[e] * HH_ + hd]);
    ebuf[idx] = v;
    atomicMax(&mbuf[dst[e] * HH_ + hd], f2ord(v));
}

// ---------------- edge pass B: ex = exp(e - m[dst]); atomic add into s[dst] -------------
__global__ void edge_b_kernel(const int* __restrict__ dst, float* __restrict__ ebuf,
                              const unsigned* __restrict__ mbuf, float* __restrict__ sbuf) {
    int idx = blockIdx.x * blockDim.x + threadIdx.x;
    if (idx >= EE_ * HH_) return;
    int e = idx >> 1, hd = idx & 1;
    float ex = __expf(ebuf[idx] - ord2f(mbuf[dst[e] * HH_ + hd]));
    ebuf[idx] = ex;
    atomicAdd(&sbuf[dst[e] * HH_ + hd], ex);
}

// ---------------- edge pass C: agg[dst] += h[src] * ex/(s[dst]+1e-9), float4 gather -----
__global__ void edge_scatter_kernel(const int* __restrict__ src, const int* __restrict__ dst,
                                    const float* __restrict__ ebuf, const float* __restrict__ sbuf,
                                    const float* __restrict__ h, float* __restrict__ agg) {
    int idx = blockIdx.x * blockDim.x + threadIdx.x;   // E*H*(C/4) = 40,960,000
    if (idx >= EE_ * HH_ * (CC_ / 4)) return;
    int c4 = idx & (CC_ / 4 - 1);
    int eh = idx >> 5;        // /(C/4)
    int hd = eh & 1;
    int e  = eh >> 1;
    int dn = dst[e], sn = src[e];
    float a = ebuf[eh] / (sbuf[dn * HH_ + hd] + 1e-9f);
    const float4 hv = *(const float4*)(h + (size_t)sn * DD_ + hd * CC_ + c4 * 4);
    float* dp = agg + (size_t)dn * DD_ + hd * CC_ + c4 * 4;
    atomicAdd(dp + 0, hv.x * a);
    atomicAdd(dp + 1, hv.y * a);
    atomicAdd(dp + 2, hv.z * a);
    atomicAdd(dp + 3, hv.w * a);
}

// ---------------- graph node count ----------------
__global__ void count_kernel(const int* __restrict__ gids, float* __restrict__ cnt) {
    int i = blockIdx.x * blockDim.x + threadIdx.x;
    if (i < NN_) atomicAdd(&cnt[gids[i]], 1.0f);
}

// ---------------- batch + graph partial stats (sum, sumsq) -----------------------------
__global__ void bstats_kernel(const float* __restrict__ y, const int* __restrict__ gids,
                              float* __restrict__ bsum, float* __restrict__ bsq,
                              float* __restrict__ gsum, float* __restrict__ gsq) {
    int d = threadIdx.x;
    int r0 = blockIdx.x * 50;
    float bs = 0.f, bq = 0.f, gs = 0.f, gq = 0.f;
    int cur = -1;
    for (int r = 0; r < 50; ++r) {
        int row = r0 + r;
        float v = y[(size_t)row * DD_ + d];
        int g = gids[row];
        if (g != cur) {
            if (cur >= 0) { atomicAdd(&gsum[cur * DD_ + d], gs); atomicAdd(&gsq[cur * DD_ + d], gq); }
            cur = g; gs = 0.f; gq = 0.f;
        }
        gs += v; gq += v * v; bs += v; bq += v * v;
    }
    if (cur >= 0) { atomicAdd(&gsum[cur * DD_ + d], gs); atomicAdd(&gsq[cur * DD_ + d], gq); }
    atomicAdd(&bsum[d], bs);
    atomicAdd(&bsq[d], bq);
}

// ---------------- finalize batch/graph mean + rstd --------------------------------------
__global__ void finalize_stats_kernel(const float* __restrict__ bsum, const float* __restrict__ bsq,
                                      const float* __restrict__ gsum, const float* __restrict__ gsq,
                                      const float* __restrict__ cnt,
                                      float* __restrict__ bmean, float* __restrict__ brstd,
                                      float* __restrict__ gmean, float* __restrict__ grstd) {
    int idx = blockIdx.x * blockDim.x + threadIdx.x;
    if (idx >= GG_ * DD_) return;
    int g = idx / DD_;
    float c = cnt[g];
    float gm = gsum[idx] / c;
    float gv = gsq[idx] / c - gm * gm;
    gmean[idx] = gm;
    grstd[idx] = rsqrtf(gv + EPS_);
    if (idx < DD_) {
        float bm = bsum[idx] / (float)NN_;
        float bv = bsq[idx] / (float)NN_ - bm * bm;
        bmean[idx] = bm;
        brstd[idx] = rsqrtf(bv + EPS_);
    }
}

// ---------------- united norm + leaky (+ f16 shadow write, optional readout) ------------
__global__ __launch_bounds__(DD_) void norm_kernel(
    const float* __restrict__ y, const int* __restrict__ gids,
    const float* __restrict__ bmean, const float* __restrict__ brstd,
    const float* __restrict__ gmean, const float* __restrict__ grstd,
    const float* __restrict__ gamma, const float* __restrict__ beta,
    const float* __restrict__ lbn, const float* __restrict__ lgn, const float* __restrict__ lnn,
    float* __restrict__ xout, _Float16* __restrict__ xh,
    float* __restrict__ embsum, int do_readout) {
    __shared__ float red[DD_];
    int n = blockIdx.x, d = threadIdx.x;
    int g = gids[n];
    float v = y[(size_t)n * DD_ + d];
    red[d] = v; __syncthreads();
    for (int s = DD_ / 2; s > 0; s >>= 1) { if (d < s) red[d] += red[d + s]; __syncthreads(); }
    float rmean = red[0] * (1.0f / DD_);
    __syncthreads();
    float dv = v - rmean;
    red[d] = dv * dv; __syncthreads();
    for (int s = DD_ / 2; s > 0; s >>= 1) { if (d < s) red[d] += red[d + s]; __syncthreads(); }
    float rvar = red[0] * (1.0f / DD_);
    float nnv = dv * rsqrtf(rvar + EPS_);
    float bnv = (v - bmean[d]) * brstd[d];
    float gnv = (v - gmean[g * DD_ + d]) * grstd[g * DD_ + d];
    float val = gamma[d] * (lbn[d] * bnv + lgn[d] * gnv + lnn[d] * nnv) + beta[d];
    val = lrelu(val);
    xout[(size_t)n * DD_ + d] = val;
    xh[(size_t)n * DD_ + d] = (_Float16)val;      // f16 shadow for next layer's WMMA
    if (do_readout) {
        __syncthreads();
        red[d] = val; __syncthreads();
        if (d < CC_) atomicAdd(&embsum[g * CC_ + d], (red[d] + red[d + CC_]) * 0.5f);
    }
}

// ---------------- final output: leaky(embsum / cnt) -------------------------------------
__global__ void final_out_kernel(const float* __restrict__ embsum, const float* __restrict__ cnt,
                                 float* __restrict__ out) {
    int idx = blockIdx.x * blockDim.x + threadIdx.x;
    if (idx >= GG_ * CC_) return;
    out[idx] = lrelu(embsum[idx] / cnt[idx / CC_]);
}

// ========================================================================================
extern "C" void kernel_launch(void* const* d_in, const int* in_sizes, int n_in,
                              void* d_out, int out_size, void* d_ws, size_t ws_size,
                              hipStream_t stream) {
    const float* node_feats = (const float*)d_in[0];
    const int*   src        = (const int*)d_in[1];
    const int*   dst        = (const int*)d_in[2];
    const int*   gids       = (const int*)d_in[3];
    const float* Wmat[3]    = { (const float*)d_in[4], (const float*)d_in[5], (const float*)d_in[6] };
    const float* resW0      = (const float*)d_in[7];
    const float *al[3], *ar[3], *gam[3], *bet[3], *lbn[3], *lgn[3], *lnn[3];
    for (int i = 0; i < 3; ++i) {
        int b = 8 + i * 7;
        al[i]  = (const float*)d_in[b + 0];
        ar[i]  = (const float*)d_in[b + 1];
        gam[i] = (const float*)d_in[b + 2];
        bet[i] = (const float*)d_in[b + 3];
        lbn[i] = (const float*)d_in[b + 4];
        lgn[i] = (const float*)d_in[b + 5];
        lnn[i] = (const float*)d_in[b + 6];
    }

    // workspace carve
    float* ws = (float*)d_ws;
    const size_t ND = (size_t)NN_ * DD_;
    float* xbuf  = ws;                      // N*D
    float* hbuf  = xbuf + ND;               // N*D
    float* agg   = hbuf + ND;               // N*D
    float* ebuf  = agg + ND;                // E*H
    float* el    = ebuf + (size_t)EE_ * HH_;// N*H
    float* er    = el + (size_t)NN_ * HH_;  // N*H
    float* mbuf  = er + (size_t)NN_ * HH_;  // N*H (ordered-uint)
    float* sbuf  = mbuf + (size_t)NN_ * HH_;// N*H
    float* bsum  = sbuf + (size_t)NN_ * HH_;// D
    float* bsq   = bsum + DD_;              // D
    float* gsum  = bsq + DD_;               // G*D
    float* gsq   = gsum + (size_t)GG_ * DD_;// G*D
    float* bmean = gsq + (size_t)GG_ * DD_; // D
    float* brstd = bmean + DD_;             // D
    float* gmean = brstd + DD_;             // G*D
    float* grstd = gmean + (size_t)GG_ * DD_;// G*D
    float* cnt   = grstd + (size_t)GG_ * DD_;// G
    float* emb   = cnt + GG_;               // G*C
    // f16 region (after float region)
    _Float16* xh    = (_Float16*)(emb + (size_t)GG_ * CC_);   // N*D halves
    _Float16* Wp[3];
    Wp[0] = xh + ND;                        // C*D
    Wp[1] = Wp[0] + (size_t)CC_ * DD_;      // D*D
    Wp[2] = Wp[1] + (size_t)DD_ * DD_;      // D*D
    _Float16* resWp = Wp[2] + (size_t)DD_ * DD_;  // C*D

    const int TB = 256;
    auto blocks = [](long long n, int tb) { return (unsigned)((n + tb - 1) / tb); };

    // graph node counts
    fill_u32_kernel<<<blocks(GG_, TB), TB, 0, stream>>>((unsigned*)cnt, 0u, GG_);
    count_kernel<<<blocks(NN_, TB), TB, 0, stream>>>(gids, cnt);

    // pack weights into WMMA fragment layout (f16), and node_feats into f16 shadow
    pack_b_kernel<<<blocks((long long)CC_ * DD_, TB), TB, 0, stream>>>(Wmat[0], Wp[0], CC_, DD_);
    pack_b_kernel<<<blocks((long long)DD_ * DD_, TB), TB, 0, stream>>>(Wmat[1], Wp[1], DD_, DD_);
    pack_b_kernel<<<blocks((long long)DD_ * DD_, TB), TB, 0, stream>>>(Wmat[2], Wp[2], DD_, DD_);
    pack_b_kernel<<<blocks((long long)CC_ * DD_, TB), TB, 0, stream>>>(resW0, resWp, CC_, DD_);
    pack_a_kernel<<<blocks((long long)NN_ * CC_, TB), TB, 0, stream>>>(node_feats, xh, NN_ * CC_);

    for (int i = 0; i < 3; ++i) {
        int K = (i == 0) ? CC_ : DD_;
        int waves = (NN_ / 16) * (DD_ / 64);

        // h = x @ W_i  (WMMA, packed f16 operands)
        gemm_wmma_f16p<<<blocks((long long)waves * 32, TB), TB, 0, stream>>>(
            xh, Wp[i], hbuf, NN_, K, DD_);

        // agg init = residual
        if (i == 0) {
            gemm_wmma_f16p<<<blocks((long long)waves * 32, TB), TB, 0, stream>>>(
                xh, resWp, agg, NN_, CC_, DD_);
        } else {
            copy_kernel<<<blocks(ND, TB), TB, 0, stream>>>(agg, xbuf, (int)ND);
        }

        // per-node-head attention logits
        attn_kernel<<<blocks((long long)NN_ * HH_ * 32, TB), TB, 0, stream>>>(hbuf, al[i], ar[i], el, er);

        // edge softmax
        fill_u32_kernel<<<blocks((long long)2 * NN_ * HH_, TB), TB, 0, stream>>>(
            (unsigned*)mbuf, 0u, 2 * NN_ * HH_);
        edge_a_kernel<<<blocks((long long)EE_ * HH_, TB), TB, 0, stream>>>(
            src, dst, el, er, ebuf, (unsigned*)mbuf);
        edge_b_kernel<<<blocks((long long)EE_ * HH_, TB), TB, 0, stream>>>(
            dst, ebuf, (const unsigned*)mbuf, sbuf);
        edge_scatter_kernel<<<blocks((long long)EE_ * HH_ * (CC_ / 4), TB), TB, 0, stream>>>(
            src, dst, ebuf, sbuf, hbuf, agg);

        // united norm stats
        int statN = 2 * DD_ + 2 * GG_ * DD_;
        fill_u32_kernel<<<blocks(statN, TB), TB, 0, stream>>>((unsigned*)bsum, 0u, statN);
        bstats_kernel<<<NN_ / 50, DD_, 0, stream>>>(agg, gids, bsum, bsq, gsum, gsq);
        finalize_stats_kernel<<<blocks((long long)GG_ * DD_, TB), TB, 0, stream>>>(
            bsum, bsq, gsum, gsq, cnt, bmean, brstd, gmean, grstd);

        int do_readout = (i == 2) ? 1 : 0;
        if (do_readout)
            fill_u32_kernel<<<blocks((long long)GG_ * CC_, TB), TB, 0, stream>>>(
                (unsigned*)emb, 0u, GG_ * CC_);
        norm_kernel<<<NN_, DD_, 0, stream>>>(agg, gids, bmean, brstd, gmean, grstd,
                                             gam[i], bet[i], lbn[i], lgn[i], lnn[i],
                                             xbuf, xh, emb, do_readout);
    }

    final_out_kernel<<<blocks((long long)GG_ * CC_, TB), TB, 0, stream>>>(emb, cnt, (float*)d_out);
}
